// MultiHeadAttention_27127013441894
// MI455X (gfx1250) — compile-verified, matrix operand order checked by reading, not run on hardware
//
#include <hip/hip_runtime.h>
#include <hip/hip_bf16.h>
#include <math.h>

// ---------------------------------------------------------------------------
// MI455X / gfx1250 multi-head attention with RoPE.
//  - all GEMM-shaped work on v_wmma_f32_16x16x32_bf16 (wave32)
//  - W tiles staged via global_load_async_to_lds_b128 (ASYNCcnt, double-buffered)
//  - flash-attention streaming softmax (no (B,H,S,S) score tensor)
//  - fragments pre-loaded in batches so WMMAs issue back-to-back after one wait
// ---------------------------------------------------------------------------

typedef __bf16 bf16_t;
typedef bf16_t v16bf __attribute__((ext_vector_type(16)));
typedef bf16_t v8bf  __attribute__((ext_vector_type(8)));
typedef float  v8f   __attribute__((ext_vector_type(8)));

#define BATCH   2
#define S_LEN   2048
#define NHEADS  16
#define HDIM    64
#define DMODEL  1024
#define NEG_BIG (-1e30f)

union FragU { v16bf v; v8bf h[2]; };

// 16x32 bf16 fragment (A-layout; B uses same per-lane map on the stored B^T):
// lane L covers row (base + L%16); elems 0..7 = k0+8*(L/16).. ; 8..15 = +16.
__device__ __forceinline__ v16bf load_frag(const bf16_t* base, int row, int ld,
                                           int k0, int lane) {
  const int hl = lane >> 4;
  const bf16_t* p = base + (size_t)row * ld + k0 + 8 * hl;
  FragU u;
  u.h[0] = *(const v8bf*)(p);
  u.h[1] = *(const v8bf*)(p + 16);
  return u.v;
}

__device__ __forceinline__ v8f wmma_bf16(v16bf a, v16bf b, v8f c) {
  return __builtin_amdgcn_wmma_f32_16x16x32_bf16(false, a, false, b,
                                                 (short)0, c, false, false);
}

__device__ __forceinline__ float rmax16(float v) {
  v = fmaxf(v, __shfl_xor(v, 1));
  v = fmaxf(v, __shfl_xor(v, 2));
  v = fmaxf(v, __shfl_xor(v, 4));
  v = fmaxf(v, __shfl_xor(v, 8));
  return v;
}
__device__ __forceinline__ float rsum16(float v) {
  v += __shfl_xor(v, 1);
  v += __shfl_xor(v, 2);
  v += __shfl_xor(v, 4);
  v += __shfl_xor(v, 8);
  return v;
}

// ---------------------------------------------------------------------------
// fp32 -> bf16 elementwise
// ---------------------------------------------------------------------------
__global__ void cvt_bf16(const float* __restrict__ src, bf16_t* __restrict__ dst,
                         int n) {
  int i = blockIdx.x * blockDim.x + threadIdx.x;
  if (i < n) dst[i] = (bf16_t)src[i];
}

// ---------------------------------------------------------------------------
// C(MxN, fp32) = A(MxK, bf16 row-major) @ W(NxK, bf16 row-major)^T
// 4 waves / block, each wave owns a 16x64 C tile. The shared 64x32 W tile is
// staged global->LDS with async-to-LDS copies (double-buffered): the copy of
// tile k+1 overlaps the 16 WMMAs on tile k; one barrier per k-step.
// ---------------------------------------------------------------------------
__global__ __launch_bounds__(128)
void gemm_xwT_bf16(const bf16_t* __restrict__ A, const bf16_t* __restrict__ W,
                   float* __restrict__ C, int M, int N, int K) {
  __shared__ bf16_t Wt[2][64 * 32];   // two 4KB W tiles (64 rows x 32 k)

  const int tid    = threadIdx.x;
  const int lane   = tid & 31;
  const int wave   = tid >> 5;
  const int lane16 = lane & 15;
  const int hl     = lane >> 4;
  const int m0 = blockIdx.y * 64 + wave * 16;
  const int n0 = blockIdx.x * 64;

  // Issue async copy of W[n0..n0+64) x [k0..k0+32) into Wt[buf].
  // 128 threads x 2 x 16B = 4KB. GVS mode: saddr = W, per-lane vgpr offset.
  auto issue_wtile = [&](int buf, int k0) {
    unsigned ldsBase = (unsigned)(uintptr_t)(&Wt[buf][0]);
#pragma unroll
    for (int it = 0; it < 2; ++it) {
      int o   = tid * 16 + it * 2048;          // byte offset within 4KB tile
      int row = o >> 6;                        // 64 bytes per tile row
      int kb  = o & 63;
      unsigned goff = (unsigned)(((n0 + row) * K + k0) * 2 + kb);
      unsigned ldsv = ldsBase + (unsigned)o;
      asm volatile("global_load_async_to_lds_b128 %0, %1, %2"
                   :: "v"(ldsv), "v"(goff), "s"(W) : "memory");
    }
  };

  const v8f vz = {0.f, 0.f, 0.f, 0.f, 0.f, 0.f, 0.f, 0.f};
  v8f acc[4] = {vz, vz, vz, vz};

  const int steps = K / 32;
  issue_wtile(0, 0);

  for (int i = 0; i < steps; ++i) {
    // Wait for this wave's outstanding async copies, then publish to all
    // waves (barrier also guarantees the buffer we are about to overwrite
    // was fully consumed in the previous step).
    asm volatile("s_wait_asynccnt 0x0" ::: "memory");
    __syncthreads();
    if (i + 1 < steps) issue_wtile((i + 1) & 1, (i + 1) * 32);

    const bf16_t* Wtile = &Wt[i & 1][0];
    // Pre-load A + all four B fragments (independent registers) so the four
    // WMMAs issue back-to-back after a single dscnt wait.
    v16bf a  = load_frag(A, m0 + lane16, K, i * 32, lane);
    v16bf b0 = load_frag(Wtile,  0 + lane16, 32, 0, lane);
    v16bf b1 = load_frag(Wtile, 16 + lane16, 32, 0, lane);
    v16bf b2 = load_frag(Wtile, 32 + lane16, 32, 0, lane);
    v16bf b3 = load_frag(Wtile, 48 + lane16, 32, 0, lane);
    acc[0] = wmma_bf16(a, b0, acc[0]);
    acc[1] = wmma_bf16(a, b1, acc[1]);
    acc[2] = wmma_bf16(a, b2, acc[2]);
    acc[3] = wmma_bf16(a, b3, acc[3]);
  }

#pragma unroll
  for (int t = 0; t < 4; ++t)
#pragma unroll
    for (int r = 0; r < 8; ++r)
      C[(size_t)(m0 + r + 8 * hl) * N + n0 + t * 16 + lane16] = acc[t][r];
}

// ---------------------------------------------------------------------------
// RoPE + pack: src fp32 (B*S, D) -> dst bf16 (B,H,S,hd); scale folded into Q.
// ---------------------------------------------------------------------------
__global__ void rope_pack_qk(const float* __restrict__ src,
                             bf16_t* __restrict__ dst, float scale) {
  int tid = blockIdx.x * blockDim.x + threadIdx.x;
  int i = tid & 31;
  int h = (tid >> 5) & (NHEADS - 1);
  int s = (tid >> 9) & (S_LEN - 1);
  int b = tid >> 20;
  if (b >= BATCH) return;

  float inv = __expf(-((float)(2 * i) / (float)HDIM) * 9.210340371976184f);
  float th = (float)s * inv;
  float sn, cs;
  __sincosf(th, &sn, &cs);

  size_t row = ((size_t)b * S_LEN + s) * DMODEL + (size_t)h * HDIM;
  float x1 = src[row + i];
  float x2 = src[row + 32 + i];
  size_t o = (((size_t)b * NHEADS + h) * S_LEN + s) * HDIM;
  dst[o + i]      = (bf16_t)((x1 * cs - x2 * sn) * scale);
  dst[o + 32 + i] = (bf16_t)((x2 * cs + x1 * sn) * scale);
}

// ---------------------------------------------------------------------------
// Pack V transposed: src fp32 (B*S, D) -> dst bf16 (B,H,hd,S).
// ---------------------------------------------------------------------------
__global__ void pack_v(const float* __restrict__ src, bf16_t* __restrict__ dst) {
  int tid = blockIdx.x * blockDim.x + threadIdx.x;
  int d = tid & (HDIM - 1);
  int h = (tid >> 6) & (NHEADS - 1);
  int s = (tid >> 10) & (S_LEN - 1);
  int b = tid >> 21;
  if (b >= BATCH) return;
  dst[(((size_t)b * NHEADS + h) * HDIM + d) * S_LEN + s] =
      (bf16_t)src[((size_t)b * S_LEN + s) * DMODEL + (size_t)h * HDIM + d];
}

// ---------------------------------------------------------------------------
// Flash attention. Grid: (S/64, B*H), 128 threads = 4 waves; each wave owns a
// 16-query tile. Per 32-key step: 4 WMMAs QK^T, online softmax via shfl_xor,
// probs C-layout -> A-layout through LDS (same-wave, s_wait_dscnt only),
// 4 WMMAs PV. Output bf16 in (B,S,H*hd).
// ---------------------------------------------------------------------------
__global__ __launch_bounds__(128)
void flash_attn_kernel(const bf16_t* __restrict__ Qb, const bf16_t* __restrict__ Kb,
                       const bf16_t* __restrict__ Vt, const int* __restrict__ mask,
                       bf16_t* __restrict__ Ob) {
  const int lane   = threadIdx.x & 31;
  const int wave   = threadIdx.x >> 5;
  const int lane16 = lane & 15;
  const int hl     = lane >> 4;
  const int bh = blockIdx.y;
  const int b  = bh >> 4;
  const int h  = bh & (NHEADS - 1);
  const int q0 = blockIdx.x * 64 + wave * 16;

  __shared__ bf16_t Psh[4][16 * 32];
  bf16_t* P = Psh[wave];

  const bf16_t* Q  = Qb + (size_t)bh * S_LEN * HDIM;
  const bf16_t* Kp = Kb + (size_t)bh * S_LEN * HDIM;
  const bf16_t* Vp = Vt + (size_t)bh * HDIM * S_LEN;
  const int* mbase = mask + ((size_t)b * S_LEN + q0) * S_LEN;

  const v16bf aq0 = load_frag(Q, q0 + lane16, HDIM, 0, lane);
  const v16bf aq1 = load_frag(Q, q0 + lane16, HDIM, 32, lane);

  const v8f vz = {0.f, 0.f, 0.f, 0.f, 0.f, 0.f, 0.f, 0.f};
  v8f acc[4] = {vz, vz, vz, vz};
  float mr[8], lr[8];
#pragma unroll
  for (int r = 0; r < 8; ++r) { mr[r] = NEG_BIG; lr[r] = 0.f; }

  for (int j0 = 0; j0 < S_LEN; j0 += 32) {
    // ---- scores: pre-load all four K fragments, then 4 WMMAs -------------
    v8f s0 = vz, s1 = vz;
    {
      v16bf b0 = load_frag(Kp, j0 + lane16, HDIM, 0, lane);
      v16bf b1 = load_frag(Kp, j0 + lane16, HDIM, 32, lane);
      v16bf c0 = load_frag(Kp, j0 + 16 + lane16, HDIM, 0, lane);
      v16bf c1 = load_frag(Kp, j0 + 16 + lane16, HDIM, 32, lane);
      s0 = wmma_bf16(aq0, b0, s0);
      s1 = wmma_bf16(aq0, c0, s1);
      s0 = wmma_bf16(aq1, b1, s0);
      s1 = wmma_bf16(aq1, c1, s1);
    }

    // ---- mask + online softmax (row m = r + 8*hl, col n = lane16) --------
#pragma unroll
    for (int r = 0; r < 8; ++r) {
      const int m = r + 8 * hl;
      const int* mptr = mbase + (size_t)m * S_LEN + j0;
      float v0 = s0[r];
      float v1 = s1[r];
      if (mptr[lane16] == 0)      v0 = NEG_BIG;
      if (mptr[16 + lane16] == 0) v1 = NEG_BIG;

      float mx   = rmax16(fmaxf(v0, v1));
      float mnew = fmaxf(mr[r], mx);
      float corr = __expf(mr[r] - mnew);
      float p0 = __expf(v0 - mnew);
      float p1 = __expf(v1 - mnew);
      float rs = rsum16(p0 + p1);
      lr[r] = lr[r] * corr + rs;
      mr[r] = mnew;
#pragma unroll
      for (int c = 0; c < 4; ++c) acc[c][r] *= corr;

      P[m * 32 + lane16]      = (bf16_t)p0;
      P[m * 32 + 16 + lane16] = (bf16_t)p1;
    }

    asm volatile("s_wait_dscnt 0" ::: "memory");

    // ---- PV: pre-load P (A-layout) + all four V fragments, then 4 WMMAs --
    v16bf ap = load_frag(P, lane16, 32, 0, lane);
    v16bf v0 = load_frag(Vp,  0 + lane16, S_LEN, j0, lane);
    v16bf v1 = load_frag(Vp, 16 + lane16, S_LEN, j0, lane);
    v16bf v2 = load_frag(Vp, 32 + lane16, S_LEN, j0, lane);
    v16bf v3 = load_frag(Vp, 48 + lane16, S_LEN, j0, lane);
    acc[0] = wmma_bf16(ap, v0, acc[0]);
    acc[1] = wmma_bf16(ap, v1, acc[1]);
    acc[2] = wmma_bf16(ap, v2, acc[2]);
    acc[3] = wmma_bf16(ap, v3, acc[3]);
  }

#pragma unroll
  for (int c = 0; c < 4; ++c)
#pragma unroll
    for (int r = 0; r < 8; ++r) {
      const int m = r + 8 * hl;
      float o = acc[c][r] / lr[r];
      Ob[((size_t)b * S_LEN + q0 + m) * DMODEL + (size_t)h * HDIM + c * 16 + lane16] =
          (bf16_t)o;
    }
}

// ---------------------------------------------------------------------------
// Driver. Inputs: x, attention_mask, Wq, Wk, Wv, Wo. Output: fp32 (B,S,D).
// ---------------------------------------------------------------------------
extern "C" void kernel_launch(void* const* d_in, const int* in_sizes, int n_in,
                              void* d_out, int out_size, void* d_ws, size_t ws_size,
                              hipStream_t stream) {
  (void)in_sizes; (void)n_in; (void)out_size; (void)ws_size;

  const float* x    = (const float*)d_in[0];
  const int*   mask = (const int*)d_in[1];
  const float* Wq   = (const float*)d_in[2];
  const float* Wk   = (const float*)d_in[3];
  const float* Wv   = (const float*)d_in[4];
  const float* Wo   = (const float*)d_in[5];

  const size_t BS = (size_t)BATCH * S_LEN;          // 4096
  const size_t XN = BS * DMODEL;                    // 4,194,304 elements
  const size_t WN = (size_t)DMODEL * DMODEL;        // 1,048,576 elements

  char* w = (char*)d_ws;
  bf16_t* xb    = (bf16_t*)w; w += XN * 2;
  bf16_t* Wqb   = (bf16_t*)w; w += WN * 2;
  bf16_t* Wkb   = (bf16_t*)w; w += WN * 2;
  bf16_t* Wvb   = (bf16_t*)w; w += WN * 2;
  bf16_t* Wob   = (bf16_t*)w; w += WN * 2;
  float*  stage = (float*)w;  w += XN * 4;
  bf16_t* Qb    = (bf16_t*)w; w += XN * 2;
  bf16_t* Kb    = (bf16_t*)w; w += XN * 2;
  bf16_t* Vt    = (bf16_t*)w; w += XN * 2;
  bf16_t* Ab    = (bf16_t*)w; w += XN * 2;          // total 64 MB

  cvt_bf16<<<(unsigned)(XN / 256), 256, 0, stream>>>(x, xb, (int)XN);
  cvt_bf16<<<(unsigned)(WN / 256), 256, 0, stream>>>(Wq, Wqb, (int)WN);
  cvt_bf16<<<(unsigned)(WN / 256), 256, 0, stream>>>(Wk, Wkb, (int)WN);
  cvt_bf16<<<(unsigned)(WN / 256), 256, 0, stream>>>(Wv, Wvb, (int)WN);
  cvt_bf16<<<(unsigned)(WN / 256), 256, 0, stream>>>(Wo, Wob, (int)WN);

  dim3 gg(DMODEL / 64, (unsigned)(BS / 64));        // (16, 64)
  const unsigned ropeBlocks = (BATCH * S_LEN * NHEADS * 32) / 256;

  gemm_xwT_bf16<<<gg, 128, 0, stream>>>(xb, Wqb, stage, (int)BS, DMODEL, DMODEL);
  rope_pack_qk<<<ropeBlocks, 256, 0, stream>>>(stage, Qb, 0.125f); // hd^-0.5
  gemm_xwT_bf16<<<gg, 128, 0, stream>>>(xb, Wkb, stage, (int)BS, DMODEL, DMODEL);
  rope_pack_qk<<<ropeBlocks, 256, 0, stream>>>(stage, Kb, 1.0f);
  gemm_xwT_bf16<<<gg, 128, 0, stream>>>(xb, Wvb, stage, (int)BS, DMODEL, DMODEL);
  pack_v<<<(unsigned)(XN / 256), 256, 0, stream>>>(stage, Vt);

  dim3 fg(S_LEN / 64, BATCH * NHEADS);              // (32, 32)
  flash_attn_kernel<<<fg, 128, 0, stream>>>(Qb, Kb, Vt, mask, Ab);

  gemm_xwT_bf16<<<gg, 128, 0, stream>>>(Ab, Wob, (float*)d_out, (int)BS, DMODEL, DMODEL);
}
